// GCN_GMT_87299505258674
// MI455X (gfx1250) — matrix operand order repaired and checked by reading, not compiled, not run on hardware
//
#include <hip/hip_runtime.h>
#include <hip/hip_bf16.h>

#define G_     256
#define N_     128
#define DEG_   16
#define NTOT_  (G_*N_)          // 32768
#define E_     (NTOT_*DEG_)     // 524288
#define EPG_   (N_*DEG_)        // 2048 edges per graph (contiguous)
#define CIN_   128
#define H_     256
#define COUT_  128
#define K1_    75
#define HEADS_ 4
#define DH_    64

typedef __attribute__((ext_vector_type(16))) __bf16 v16bf;
typedef __attribute__((ext_vector_type(8)))  float  v8f;

// ---------------------------------------------------------------------------
// Weight repack: W[K,Nc] fp32 -> Wp bf16 in WMMA-B fragment order:
//   Wp[((k0/32)*Nc + n)*32 + j] = bf16(W[(k0 + j)*Nc + n]),  j = 0..31
// so each lane's B fragment is 16 contiguous bf16 (one 32B vector load).
// ---------------------------------------------------------------------------
__global__ void pack_w(const float* __restrict__ W, __bf16* __restrict__ Wp,
                       int K, int Nc)
{
    size_t tot = (size_t)K * Nc;
    for (size_t i = blockIdx.x * (size_t)blockDim.x + threadIdx.x; i < tot;
         i += (size_t)gridDim.x * blockDim.x) {
        int    j  = (int)(i & 31);
        size_t tn = i >> 5;
        int    n  = (int)(tn % Nc);
        int    t  = (int)(tn / Nc);
        Wp[i] = (__bf16)W[(size_t)(t * 32 + j) * Nc + n];
    }
}

// ---------------------------------------------------------------------------
// GEMM: out[M,Nc] = A[M,K] @ W[K,Nc] (+bias), bf16 WMMA, fp32 accumulate.
// One wave32 per 16x64 output strip (4 accumulators): A fragment loaded once,
// 4 v_wmma_f32_16x16x32_bf16 per k-step. Compile-time specialization:
//   MODE 0: out = T      MODE 1: out = res + relu(T)
//   HAS_BIAS: bias[] add      TAIL: M not multiple of 16 (row clamp + guard)
// Only the 4 (MODE,HAS_BIAS,TAIL) combos actually launched are instantiated.
// ---------------------------------------------------------------------------
template <int MODE, bool HAS_BIAS, bool TAIL>
__global__ void gemm_bf16(const float* __restrict__ A, const __bf16* __restrict__ Wp,
                          const float* __restrict__ bias, const float* __restrict__ res,
                          float* __restrict__ out, int M, int K, int Nc)
{
    const int nt  = Nc >> 6;                    // 64-wide strips
    const int mt  = (M + 15) >> 4;
    const int wid = (int)((blockIdx.x * blockDim.x + threadIdx.x) >> 5);
    if (wid >= mt * nt) return;                 // wave-uniform branch
    const int lane = threadIdx.x & 31;
    const int tm = (wid / nt) << 4;
    const int tn = (wid % nt) << 6;
    int row = tm + (lane & 15);
    if (TAIL) row = (row < M) ? row : (M - 1);  // clamp: no divergent loads
    const int n   = tn + (lane & 15);
    const int kbA = (lane < 16) ? 0 : 16;       // A: k = 16h + s (matches B packing)
    const int kbB = (lane < 16) ? 0 : 16;       // B: k = 16h + s (contiguous, same as A)

    const float* Ar = A + (size_t)row * K;
    v8f acc[4];
#pragma unroll
    for (int j = 0; j < 4; ++j) acc[j] = (v8f){0.f,0.f,0.f,0.f,0.f,0.f,0.f,0.f};

    for (int k0 = 0; k0 < K; k0 += 32) {
        const float4* Ap = (const float4*)(Ar + k0 + kbA);  // 32B aligned
        float4 a0 = Ap[0], a1 = Ap[1];          // k = kbA + 0..7
        float4 a2 = Ap[2], a3 = Ap[3];          // k = kbA + 8..15
        v16bf a;
        a[0]=(__bf16)a0.x;  a[1]=(__bf16)a0.y;  a[2]=(__bf16)a0.z;  a[3]=(__bf16)a0.w;
        a[4]=(__bf16)a1.x;  a[5]=(__bf16)a1.y;  a[6]=(__bf16)a1.z;  a[7]=(__bf16)a1.w;
        a[8]=(__bf16)a2.x;  a[9]=(__bf16)a2.y;  a[10]=(__bf16)a2.z; a[11]=(__bf16)a2.w;
        a[12]=(__bf16)a3.x; a[13]=(__bf16)a3.y; a[14]=(__bf16)a3.z; a[15]=(__bf16)a3.w;

        const __bf16* Wb = Wp + ((size_t)(k0 >> 5) * Nc + n) * 32 + kbB;
#pragma unroll
        for (int j = 0; j < 4; ++j) {
            v16bf b = *(const v16bf*)(Wb + (size_t)j * 16 * 32);  // 32B contiguous
            acc[j] = __builtin_amdgcn_wmma_f32_16x16x32_bf16(false, a, false, b,
                                                             (short)0, acc[j],
                                                             false, false);
        }
    }

    const int mofs = (lane < 16) ? 0 : 8;       // ISA 32-bit C/D 16x16 lane map
#pragma unroll
    for (int j = 0; j < 4; ++j) {
        const int nj = n + j * 16;
        const float bv = HAS_BIAS ? bias[nj] : 0.f;
        float*       ob = out + (size_t)(tm + mofs) * Nc + nj;
        const float* rb = (MODE == 1) ? (res + (size_t)(tm + mofs) * Nc + nj) : nullptr;
#pragma unroll
        for (int i = 0; i < 8; ++i) {
            if (!TAIL || (tm + mofs + i) < M) {
                float t = acc[j][i] + bv;
                if (MODE == 1) t = rb[(size_t)i * Nc] + fmaxf(t, 0.f);
                ob[(size_t)i * Nc] = t;
            }
        }
    }
}

// ---------------------------------------------------------------------------
// GIN aggregation per graph: out[i] = x[i] + sum_{e: dst=i} x[src_e]
// ---------------------------------------------------------------------------
extern __shared__ float smem[];
__global__ void gin_agg(const float* __restrict__ x, const int* __restrict__ src,
                        const int* __restrict__ dst, float* __restrict__ out, int C)
{
    const int g  = blockIdx.x;
    const int n0 = g * N_;
    float* acc = smem;                                    // 128*C floats
    for (int i = threadIdx.x; i < N_ * C; i += blockDim.x)
        acc[i] = x[(size_t)n0 * C + i];
    __syncthreads();
    const int e0 = g * EPG_;
    for (int e = threadIdx.x; e < EPG_; e += blockDim.x) {
        const int s  = src[e0 + e];
        const int dl = dst[e0 + e] - n0;
        const float* xs = x + (size_t)s * C;
        for (int c = 0; c < C; ++c)
            atomicAdd(&acc[dl * C + c], xs[c]);
    }
    __syncthreads();
    for (int i = threadIdx.x; i < N_ * C; i += blockDim.x)
        out[(size_t)n0 * C + i] = acc[i];
}

// ---------------------------------------------------------------------------
// BatchNorm (training-mode) stats: one block per channel.
// ---------------------------------------------------------------------------
__global__ void bn_stats(const float* __restrict__ h, int M, int C,
                         float* __restrict__ mean, float* __restrict__ inv)
{
    const int c = blockIdx.x;
    float s = 0.f, s2 = 0.f;
    for (int m = threadIdx.x; m < M; m += blockDim.x) {
        float v = h[(size_t)m * C + c];
        s += v; s2 += v * v;
    }
    __shared__ float rs[256], rq[256];
    rs[threadIdx.x] = s; rq[threadIdx.x] = s2;
    __syncthreads();
    for (int o = 128; o > 0; o >>= 1) {
        if ((int)threadIdx.x < o) { rs[threadIdx.x] += rs[threadIdx.x + o];
                                    rq[threadIdx.x] += rq[threadIdx.x + o]; }
        __syncthreads();
    }
    if (threadIdx.x == 0) {
        float mu  = rs[0] / (float)M;
        float var = rq[0] / (float)M - mu * mu;
        mean[c] = mu;
        inv[c]  = rsqrtf(var + 1e-5f);
    }
}

__global__ void bn_apply(const float* __restrict__ h, float* __restrict__ out,
                         const float* __restrict__ mean, const float* __restrict__ inv,
                         const float* __restrict__ gamma, const float* __restrict__ beta,
                         int M, int C, int relu)
{
    size_t tot = (size_t)M * C;
    for (size_t i = blockIdx.x * (size_t)blockDim.x + threadIdx.x; i < tot;
         i += (size_t)gridDim.x * blockDim.x) {
        int c = (int)(i % C);
        float v = (h[i] - mean[c]) * inv[c] * gamma[c] + beta[c];
        if (relu) v = fmaxf(v, 0.f);
        out[i] = v;
    }
}

// ---------------------------------------------------------------------------
// Degrees (with self-loop +1) and rsqrt(deg): one block per graph.
// ---------------------------------------------------------------------------
__global__ void deg_kernel(const int* __restrict__ dst, float* __restrict__ deg,
                           float* __restrict__ dinv)
{
    __shared__ int cnt[N_];
    const int g = blockIdx.x;
    if (threadIdx.x < N_) cnt[threadIdx.x] = 0;
    __syncthreads();
    const int e0 = g * EPG_;
    for (int e = threadIdx.x; e < EPG_; e += blockDim.x)
        atomicAdd(&cnt[dst[e0 + e] - g * N_], 1);
    __syncthreads();
    if (threadIdx.x < N_) {
        float d = (float)cnt[threadIdx.x] + 1.0f;
        deg[g * N_ + threadIdx.x]  = d;
        dinv[g * N_ + threadIdx.x] = rsqrtf(d);
    }
}

// ---------------------------------------------------------------------------
// GCN scatter: out[i] = sum_{e:dst=i} xw[src]*dinv[src]*dinv[dst] + xw[i]/deg[i] + b
// ---------------------------------------------------------------------------
__global__ void gcn_scatter(const float* __restrict__ xw, const int* __restrict__ src,
                            const int* __restrict__ dst, const float* __restrict__ deg,
                            const float* __restrict__ dinv, const float* __restrict__ bias,
                            float* __restrict__ out)
{
    const int g  = blockIdx.x;
    const int n0 = g * N_;
    float* acc = smem;                                    // 128*H_ floats
    for (int i = threadIdx.x; i < N_ * H_; i += blockDim.x) {
        int nl = i / H_;
        acc[i] = xw[(size_t)n0 * H_ + i] / deg[n0 + nl];
    }
    __syncthreads();
    const int e0 = g * EPG_;
    for (int e = threadIdx.x; e < EPG_; e += blockDim.x) {
        const int s = src[e0 + e];
        const int d = dst[e0 + e];
        const float w = dinv[s] * dinv[d];
        const int dl = d - n0;
        const float* xs = xw + (size_t)s * H_;
        for (int c = 0; c < H_; ++c)
            atomicAdd(&acc[dl * H_ + c], xs[c] * w);
    }
    __syncthreads();
    for (int i = threadIdx.x; i < N_ * H_; i += blockDim.x)
        out[(size_t)n0 * H_ + i] = acc[i] + bias[i % H_];
}

// ---------------------------------------------------------------------------
// Multihead attention core: O[g,q,:] = Qp[g,q,:] + softmax(Qp Kd^T / 16) Vd
// One block of 128 threads per (graph, head). qstride==0 => Qp broadcast.
// Dyn LDS layout: K[kc*64] | V[kc*64] | sc[128] | qrow[64] | red[128]
// ---------------------------------------------------------------------------
__global__ void attn_kernel(const float* __restrict__ Qp, int qstride,
                            const float* __restrict__ Kd, const float* __restrict__ Vd,
                            float* __restrict__ O, int q, int kc)
{
    const int g  = blockIdx.x >> 2;
    const int hd = (blockIdx.x & 3) * DH_;
    float* Ks   = smem;
    float* Vs   = Ks + kc * DH_;
    float* sc   = Vs + kc * DH_;
    float* qrow = sc + 128;
    float* red  = qrow + DH_;
    const float* Kg = Kd + (size_t)g * kc * H_ + hd;
    const float* Vg = Vd + (size_t)g * kc * H_ + hd;
    for (int i = threadIdx.x; i < kc * DH_; i += blockDim.x) {
        int kk = i >> 6, dd = i & 63;
        Ks[i] = Kg[(size_t)kk * H_ + dd];
        Vs[i] = Vg[(size_t)kk * H_ + dd];
    }
    __syncthreads();
    const float scale = 0.0625f;                 // 1/sqrt(H=256)
    const float* Qg = Qp + (size_t)g * qstride;
    for (int qi = 0; qi < q; ++qi) {
        if (threadIdx.x < DH_)
            qrow[threadIdx.x] = Qg[(size_t)qi * H_ + hd + threadIdx.x];
        __syncthreads();
        float sv = -INFINITY;
        if ((int)threadIdx.x < kc) {
            sv = 0.f;
            for (int dd = 0; dd < DH_; ++dd)
                sv += qrow[dd] * Ks[threadIdx.x * DH_ + dd];
            sv *= scale;
        }
        sc[threadIdx.x]  = sv;
        red[threadIdx.x] = sv;
        __syncthreads();
        for (int o = 64; o > 0; o >>= 1) {
            if ((int)threadIdx.x < o)
                red[threadIdx.x] = fmaxf(red[threadIdx.x], red[threadIdx.x + o]);
            __syncthreads();
        }
        const float mx = red[0];
        __syncthreads();
        float ex = ((int)threadIdx.x < kc) ? __expf(sc[threadIdx.x] - mx) : 0.f;
        sc[threadIdx.x]  = ex;
        red[threadIdx.x] = ex;
        __syncthreads();
        for (int o = 64; o > 0; o >>= 1) {
            if ((int)threadIdx.x < o) red[threadIdx.x] += red[threadIdx.x + o];
            __syncthreads();
        }
        const float invs = 1.0f / red[0];
        __syncthreads();
        if (threadIdx.x < DH_) {
            float ov = 0.f;
            for (int kk = 0; kk < kc; ++kk)
                ov += sc[kk] * Vs[kk * DH_ + threadIdx.x];
            ov *= invs;
            size_t oidx = ((size_t)g * q + qi) * H_ + hd + threadIdx.x;
            O[oidx] = Qg[(size_t)qi * H_ + hd + threadIdx.x] + ov;
        }
        __syncthreads();
    }
}

// ---------------------------------------------------------------------------
// Only 4 GEMM variants are ever launched:
//   mode1 (+bias, M%16==0), mode0+bias (M%16==0), mode0 no-bias (M%16==0),
//   mode0+bias tail (Qp GEMMs, M=75 / M=1).
// ---------------------------------------------------------------------------
static inline void launch_gemm(const float* A, const __bf16* Wp, const float* bias,
                               const float* res, float* out, int M, int K, int Nc,
                               int mode, hipStream_t s)
{
    long waves  = (long)((M + 15) / 16) * (Nc / 64);
    dim3 gd((unsigned)((waves * 32 + 255) / 256)), bd(256);
    const bool tail = (M & 15) != 0;
    if (mode == 1)
        gemm_bf16<1, true, false><<<gd, bd, 0, s>>>(A, Wp, bias, res, out, M, K, Nc);
    else if (tail)
        gemm_bf16<0, true, true ><<<gd, bd, 0, s>>>(A, Wp, bias, res, out, M, K, Nc);
    else if (bias)
        gemm_bf16<0, true, false><<<gd, bd, 0, s>>>(A, Wp, bias, res, out, M, K, Nc);
    else
        gemm_bf16<0, false, false><<<gd, bd, 0, s>>>(A, Wp, bias, res, out, M, K, Nc);
}

extern "C" void kernel_launch(void* const* d_in, const int* in_sizes, int n_in,
                              void* d_out, int out_size, void* d_ws, size_t ws_size,
                              hipStream_t stream)
{
    (void)in_sizes; (void)n_in; (void)out_size; (void)ws_size;
    const float* x   = (const float*)d_in[0];
    const int*   ei  = (const int*)d_in[1];
    const int*   src = ei;
    const int*   dst = ei + E_;
    const float* g0_W1 = (const float*)d_in[3],  *g0_b1 = (const float*)d_in[4];
    const float* g0_g1 = (const float*)d_in[5],  *g0_be1 = (const float*)d_in[6];
    const float* g0_W2 = (const float*)d_in[7],  *g0_b2 = (const float*)d_in[8];
    const float* g0_g2 = (const float*)d_in[9],  *g0_be2 = (const float*)d_in[10];
    const float* g1_W1 = (const float*)d_in[11], *g1_b1 = (const float*)d_in[12];
    const float* g1_g1 = (const float*)d_in[13], *g1_be1 = (const float*)d_in[14];
    const float* g1_W2 = (const float*)d_in[15], *g1_b2 = (const float*)d_in[16];
    const float* g1_g2 = (const float*)d_in[17], *g1_be2 = (const float*)d_in[18];
    const float* lin1_W = (const float*)d_in[19], *lin1_b = (const float*)d_in[20];
    const float* S1 = (const float*)d_in[21];
    const float* p1_Wq=(const float*)d_in[22], *p1_Wk=(const float*)d_in[23];
    const float* p1_Wv=(const float*)d_in[24], *p1_Wo=(const float*)d_in[25];
    const float* p1_bq=(const float*)d_in[26], *p1_bk=(const float*)d_in[27];
    const float* p1_bv=(const float*)d_in[28], *p1_bo=(const float*)d_in[29];
    const float* s2_Wq=(const float*)d_in[30], *s2_Wk=(const float*)d_in[31];
    const float* s2_Wv=(const float*)d_in[32], *s2_Wo=(const float*)d_in[33];
    const float* s2_bq=(const float*)d_in[34], *s2_bk=(const float*)d_in[35];
    const float* s2_bv=(const float*)d_in[36], *s2_bo=(const float*)d_in[37];
    const float* p3_Wq=(const float*)d_in[38], *p3_Wk=(const float*)d_in[39];
    const float* p3_Wv=(const float*)d_in[40], *p3_Wo=(const float*)d_in[41];
    const float* p3_bq=(const float*)d_in[42], *p3_bk=(const float*)d_in[43];
    const float* p3_bv=(const float*)d_in[44], *p3_bo=(const float*)d_in[45];
    const float* S3 = (const float*)d_in[46];
    const float* lin2_W = (const float*)d_in[47], *lin2_b = (const float*)d_in[48];
    float* out = (float*)d_out;

    // ---- workspace carve-out (fp32 region, then bf16 packed-weight region) ----
    const size_t MN  = (size_t)NTOT_ * H_;        // node buffers
    const size_t XPN = (size_t)G_ * K1_ * H_;     // pooled buffers
    float* ws  = (float*)d_ws;
    float* B0  = ws; ws += MN;
    float* B1  = ws; ws += MN;
    float* B2  = ws; ws += MN;
    float* B3  = ws; ws += MN;
    float* XA  = ws; ws += XPN;
    float* XB  = ws; ws += XPN;
    float* XC  = ws; ws += XPN;
    float* XD  = ws; ws += XPN;
    float* Qp1 = ws; ws += 80 * H_;
    float* Qp3 = ws; ws += 16 * H_;
    float* Ob  = ws; ws += (size_t)G_ * H_;
    float* Ob2 = ws; ws += (size_t)G_ * H_;
    float* mean = ws; ws += H_;
    float* invs = ws; ws += H_;
    float* degb = ws; ws += NTOT_;
    float* dinv = ws; ws += NTOT_;

    __bf16* wp = (__bf16*)ws;                     // 64B-aligned by construction
    auto pack = [&](const float* W, int K, int Nc) -> __bf16* {
        __bf16* p = wp; wp += (size_t)K * Nc;
        pack_w<<<256, 256, 0, stream>>>(W, p, K, Nc);
        return p;
    };
    __bf16* pk_g0W1 = pack(g0_W1, CIN_, H_);
    __bf16* pk_g0W2 = pack(g0_W2, H_, H_);
    __bf16* pk_g1W1 = pack(g1_W1, H_, H_);
    __bf16* pk_g1W2 = pack(g1_W2, H_, H_);
    __bf16* pk_lin1 = pack(lin1_W, H_, H_);
    __bf16* pk_p1q  = pack(p1_Wq, H_, H_);
    __bf16* pk_p1k  = pack(p1_Wk, H_, H_);
    __bf16* pk_p1v  = pack(p1_Wv, H_, H_);
    __bf16* pk_p1o  = pack(p1_Wo, H_, H_);
    __bf16* pk_s2q  = pack(s2_Wq, H_, H_);
    __bf16* pk_s2k  = pack(s2_Wk, H_, H_);
    __bf16* pk_s2v  = pack(s2_Wv, H_, H_);
    __bf16* pk_s2o  = pack(s2_Wo, H_, H_);
    __bf16* pk_p3q  = pack(p3_Wq, H_, H_);
    __bf16* pk_p3k  = pack(p3_Wk, H_, H_);
    __bf16* pk_p3v  = pack(p3_Wv, H_, H_);
    __bf16* pk_p3o  = pack(p3_Wo, H_, H_);
    __bf16* pk_lin2 = pack(lin2_W, H_, COUT_);

    // ---- GIN layer 0 ----
    gin_agg<<<G_, 256, N_ * CIN_ * 4, stream>>>(x, src, dst, B0, CIN_);
    launch_gemm(B0, pk_g0W1, g0_b1, nullptr, B1, NTOT_, CIN_, H_, 0, stream);
    bn_stats<<<H_, 256, 0, stream>>>(B1, NTOT_, H_, mean, invs);
    bn_apply<<<4096, 256, 0, stream>>>(B1, B2, mean, invs, g0_g1, g0_be1, NTOT_, H_, 1);
    launch_gemm(B2, pk_g0W2, g0_b2, nullptr, B1, NTOT_, H_, H_, 0, stream);
    bn_stats<<<H_, 256, 0, stream>>>(B1, NTOT_, H_, mean, invs);
    bn_apply<<<4096, 256, 0, stream>>>(B1, B0, mean, invs, g0_g2, g0_be2, NTOT_, H_, 1);

    // ---- GIN layer 1 ----
    gin_agg<<<G_, 256, N_ * H_ * 4, stream>>>(B0, src, dst, B2, H_);
    launch_gemm(B2, pk_g1W1, g1_b1, nullptr, B1, NTOT_, H_, H_, 0, stream);
    bn_stats<<<H_, 256, 0, stream>>>(B1, NTOT_, H_, mean, invs);
    bn_apply<<<4096, 256, 0, stream>>>(B1, B2, mean, invs, g1_g1, g1_be1, NTOT_, H_, 1);
    launch_gemm(B2, pk_g1W2, g1_b2, nullptr, B1, NTOT_, H_, H_, 0, stream);
    bn_stats<<<H_, 256, 0, stream>>>(B1, NTOT_, H_, mean, invs);
    bn_apply<<<4096, 256, 0, stream>>>(B1, B3, mean, invs, g1_g2, g1_be2, NTOT_, H_, 1);

    // ---- lin1: xl ----
    launch_gemm(B3, pk_lin1, lin1_b, nullptr, B0, NTOT_, H_, H_, 0, stream);

    // ---- GCN K/V for GMPool_G ----
    deg_kernel<<<G_, 256, 0, stream>>>(dst, degb, dinv);
    launch_gemm(B0, pk_p1k, nullptr, nullptr, B1, NTOT_, H_, H_, 0, stream);
    gcn_scatter<<<G_, 256, N_ * H_ * 4, stream>>>(B1, src, dst, degb, dinv, p1_bk, B2);
    launch_gemm(B0, pk_p1v, nullptr, nullptr, B1, NTOT_, H_, H_, 0, stream);
    gcn_scatter<<<G_, 256, N_ * H_ * 4, stream>>>(B1, src, dst, degb, dinv, p1_bv, B3);

    // ---- GMPool_G: PMA (K1=75 seeds, broadcast Qp) ----
    launch_gemm(S1, pk_p1q, p1_bq, nullptr, Qp1, K1_, H_, H_, 0, stream);
    attn_kernel<<<G_ * HEADS_, 128, (N_ * 128 + 320) * 4, stream>>>(
        Qp1, 0, B2, B3, XA, K1_, N_);
    launch_gemm(XA, pk_p1o, p1_bo, XA, XB, G_ * K1_, H_, H_, 1, stream);

    // ---- SelfAtt (SAB) ----
    launch_gemm(XB, pk_s2q, s2_bq, nullptr, XD, G_ * K1_, H_, H_, 0, stream);
    launch_gemm(XB, pk_s2k, s2_bk, nullptr, XA, G_ * K1_, H_, H_, 0, stream);
    launch_gemm(XB, pk_s2v, s2_bv, nullptr, XC, G_ * K1_, H_, H_, 0, stream);
    attn_kernel<<<G_ * HEADS_, 128, (K1_ * 128 + 320) * 4, stream>>>(
        XD, K1_ * H_, XA, XC, XB, K1_, K1_);
    launch_gemm(XB, pk_s2o, s2_bo, XB, XA, G_ * K1_, H_, H_, 1, stream);

    // ---- GMPool_I: PMA with single seed ----
    launch_gemm(XA, pk_p3k, p3_bk, nullptr, XB, G_ * K1_, H_, H_, 0, stream);
    launch_gemm(XA, pk_p3v, p3_bv, nullptr, XC, G_ * K1_, H_, H_, 0, stream);
    launch_gemm(S3, pk_p3q, p3_bq, nullptr, Qp3, 1, H_, H_, 0, stream);
    attn_kernel<<<G_ * HEADS_, 128, (K1_ * 128 + 320) * 4, stream>>>(
        Qp3, 0, XB, XC, Ob, 1, K1_);
    launch_gemm(Ob, pk_p3o, p3_bo, Ob, Ob2, G_, H_, H_, 1, stream);

    // ---- lin2: final classifier ----
    launch_gemm(Ob2, pk_lin2, lin2_b, nullptr, out, G_, H_, COUT_, 0, stream);
}
